// MultiHeadAttention_16054587752737
// MI455X (gfx1250) — matrix operand order repaired and checked by reading, not compiled, not run on hardware
//
#include <hip/hip_runtime.h>
#include <hip/hip_bf16.h>
#include <stdint.h>

// ---------------------------------------------------------------------------
// MI455X (gfx1250) fused multi-head attention
//   wave32 + v_wmma_f32_16x16x32_f16 + global_load_async_to_lds_b128
// ---------------------------------------------------------------------------

typedef __attribute__((ext_vector_type(16))) _Float16 v16h;
typedef __attribute__((ext_vector_type(8)))  _Float16 v8h;
typedef __attribute__((ext_vector_type(4)))  _Float16 v4h;
typedef __attribute__((ext_vector_type(8)))  float    v8f;
typedef __attribute__((ext_vector_type(4)))  float    v4f;

#define N_EMBD   384
#define NHEAD    6
#define HSIZE    64
#define BB       128
#define TT       256
#define M_TOTAL  (BB * TT)        /* 32768 token rows */
#define QKV_N    (3 * N_EMBD)     /* 1152: [Q | K | V] packed per row */

// LDS strides (halves): padded against bank conflicts, rows stay 16B-aligned
#define LDQ   72
#define LDS_S 264
#define LDVT  264

// ---------------------------------------------------------------------------
// WMMA fragment loaders per CDNA5 ISA VGPR layouts (CDNA5 ISA):
//   A (16x32 f16): lane -> row M=lane%16; halves 0..7 = K[8*hi..], 8..15 = K[16+8*hi..]
//   B (32x16 f16): lane -> col N=lane%16; halves 0..7 = K[8*hi..], 8..15 = K[16+8*hi..]
// B is always fed from N-major storage so lanes issue contiguous 16B loads.
// ---------------------------------------------------------------------------

__device__ inline v16h load_a_h16(const _Float16* __restrict__ rowK0, int hi) {
    union { v16h v; v8h p[2]; } a;
    a.p[0] = *(const v8h*)(rowK0 + 8 * hi);
    a.p[1] = *(const v8h*)(rowK0 + 16 + 8 * hi);
    return a.v;
}

__device__ inline v16h load_b_h16(const _Float16* __restrict__ nrowK0, int hi) {
    union { v16h v; v8h p[2]; } b;
    const _Float16* p = nrowK0 + 8 * hi;
    b.p[0] = *(const v8h*)(p);
    b.p[1] = *(const v8h*)(p + 16);
    return b.v;
}

__device__ inline v8f wmma_f16(v16h a, v16h b, v8f c) {
    return __builtin_amdgcn_wmma_f32_16x16x32_f16(
        false, a, false, b, (short)0, c, false, false);
}

// ---------------------------------------------------------------------------
// Kernel 0: one-shot fp32 -> fp16 conversion of x and all weights, hoisting
// every v_cvt out of the GEMM hot loops.  w16 packs [Wq;Wk;Wv] as 1152 rows.
// ---------------------------------------------------------------------------
#define NX  ((size_t)M_TOTAL * N_EMBD)        /* 12582912 */
#define NW1 ((size_t)N_EMBD * N_EMBD)         /* 147456 per weight */
#define NW  (3 * NW1)                         /* 442368 */

__global__ __launch_bounds__(256) void cvt_kernel(
    const float* __restrict__ x,
    const float* __restrict__ Wq, const float* __restrict__ Wk,
    const float* __restrict__ Wv, const float* __restrict__ Wo,
    _Float16* __restrict__ x16, _Float16* __restrict__ w16,
    _Float16* __restrict__ wo16) {

    const size_t i = ((size_t)blockIdx.x * 256 + threadIdx.x) * 4;
    const float* src;
    _Float16* dst;
    size_t j;
    if (i < NX)           { src = x;  dst = x16;  j = i; }
    else if (i < NX + NW) {
        j = i - NX;
        src = (j < NW1) ? Wq : (j < 2 * NW1 ? Wk : Wv);
        dst = w16;
        src -= (j < NW1) ? 0 : (j < 2 * NW1 ? NW1 : 2 * NW1);
    } else                { src = Wo; dst = wo16; j = i - NX - NW; }

    v4f f = *(const v4f*)(src + j);
    v4h h;
#pragma unroll
    for (int u = 0; u < 4; ++u) h[u] = (_Float16)f[u];
    *(v4h*)(dst + j) = h;
}

// ---------------------------------------------------------------------------
// Kernel 1: QKV projection, pure-fp16 operands.  Each wave owns a 16x64 tile:
// one A fragment feeds 4 WMMAs per k-step (48 WMMAs / wave, K fully unrolled).
// ---------------------------------------------------------------------------
__global__ __launch_bounds__(256) void qkv_kernel(
    const _Float16* __restrict__ x16,
    const _Float16* __restrict__ w16,
    _Float16* __restrict__ qkv) {

    const int lane = threadIdx.x & 31, wave = threadIdx.x >> 5;
    const int lo = lane & 15, hi = lane >> 4;

    const int cGroups = QKV_N / 64;            // 18
    const int task = blockIdx.x * 8 + wave;    // 2048*18 = 36864 tasks
    const int mT = task / cGroups;
    const int cT = task - mT * cGroups;
    const int m0 = mT * 16, n0 = cT * 64;

    const _Float16* arow = x16 + (size_t)(m0 + lo) * N_EMBD;
    const _Float16* brow = w16 + (size_t)(n0 + lo) * N_EMBD;

    v8f acc0 = {}, acc1 = {}, acc2 = {}, acc3 = {};
#pragma unroll
    for (int k0 = 0; k0 < N_EMBD; k0 += 32) {
        if (k0 + 64 < N_EMBD) __builtin_prefetch(arow + k0 + 64, 0, 1);
        v16h a = load_a_h16(arow + k0, hi);
        acc0 = wmma_f16(a, load_b_h16(brow + k0, hi), acc0);
        acc1 = wmma_f16(a, load_b_h16(brow + (size_t)16 * N_EMBD + k0, hi), acc1);
        acc2 = wmma_f16(a, load_b_h16(brow + (size_t)32 * N_EMBD + k0, hi), acc2);
        acc3 = wmma_f16(a, load_b_h16(brow + (size_t)48 * N_EMBD + k0, hi), acc3);
    }

    _Float16* orow = qkv + (size_t)(m0 + 8 * hi) * QKV_N + n0 + lo;
#pragma unroll
    for (int j = 0; j < 8; ++j) {
        orow[(size_t)j * QKV_N +  0] = (_Float16)acc0[j];
        orow[(size_t)j * QKV_N + 16] = (_Float16)acc1[j];
        orow[(size_t)j * QKV_N + 32] = (_Float16)acc2[j];
        orow[(size_t)j * QKV_N + 48] = (_Float16)acc3[j];
    }
}

// ---------------------------------------------------------------------------
// Kernel 2: causal attention, one WG (8 wave32) per (b,h), 237KB LDS.
// Q/K head slices stream into LDS through the gfx1250 async-copy path
// (global_load_async_to_lds_b128 + s_wait_asynccnt); V is transposed via VGPRs.
// ---------------------------------------------------------------------------
__global__ __launch_bounds__(256) void attn_kernel(
    const _Float16* __restrict__ qkv,
    _Float16* __restrict__ attn) {

    const int b = blockIdx.x / NHEAD;
    const int h = blockIdx.x - b * NHEAD;

    extern __shared__ _Float16 smem[];
    _Float16* sQ  = smem;                                 // [256][72]
    _Float16* sK  = smem + TT * LDQ;                      // [256][72]
    _Float16* sVt = smem + 2 * TT * LDQ;                  // [64][264]
    _Float16* sS  = smem + 2 * TT * LDQ + HSIZE * LDVT;   // [256][264]

    const int t = threadIdx.x;
    const _Float16* base = qkv + (size_t)(b * TT) * QKV_N + h * HSIZE;

    // ---- async stage Q,K (16B granules, LDS dest = low 32 addr bits) ----
    {
        const uint32_t sQoff = (uint32_t)(uintptr_t)(void*)sQ;
        const uint32_t sKoff = (uint32_t)(uintptr_t)(void*)sK;
        for (int i = t; i < TT * 8; i += 256) {           // 2048 chunks each
            const int r = i >> 3, c8 = (i & 7) * 8;
            const uint64_t gq = (uint64_t)(uintptr_t)(base + (size_t)r * QKV_N + c8);
            const uint64_t gk = (uint64_t)(uintptr_t)(base + (size_t)r * QKV_N + N_EMBD + c8);
            const uint32_t lq = sQoff + (uint32_t)(r * LDQ + c8) * 2;
            const uint32_t lk = sKoff + (uint32_t)(r * LDQ + c8) * 2;
            asm volatile("global_load_async_to_lds_b128 %0, %1, off"
                         :: "v"(lq), "v"(gq) : "memory");
            asm volatile("global_load_async_to_lds_b128 %0, %1, off"
                         :: "v"(lk), "v"(gk) : "memory");
        }
        // V^T staged through VGPRs (transpose on LDS write)
        for (int i = t; i < TT * 8; i += 256) {
            const int r = i >> 3, c8 = (i & 7) * 8;
            v8h v = *(const v8h*)(base + (size_t)r * QKV_N + 2 * N_EMBD + c8);
#pragma unroll
            for (int u = 0; u < 8; ++u) sVt[(c8 + u) * LDVT + r] = v[u];
        }
        asm volatile("s_wait_asynccnt 0" ::: "memory");
    }
    __syncthreads();

    const int lane = t & 31, wave = t >> 5;
    const int lo = lane & 15, hi = lane >> 4;

    // ---- S = Q K^T: wave owns row-tiles {2w,2w+1}; A (K=64) lives in regs;
    //      strictly-upper causal tiles are skipped at WMMA granularity ----
    for (int mT = wave * 2; mT < wave * 2 + 2; ++mT) {
        const _Float16* arow = sQ + (mT * 16 + lo) * LDQ;
        v16h a0 = load_a_h16(arow, hi);
        v16h a1 = load_a_h16(arow + 32, hi);
        for (int nT = 0; nT <= mT; ++nT) {
            const _Float16* brow = sK + (nT * 16 + lo) * LDQ;
            v8f acc = {};
            acc = wmma_f16(a0, load_b_h16(brow, hi), acc);
            acc = wmma_f16(a1, load_b_h16(brow + 32, hi), acc);
            _Float16* srow = sS + (mT * 16 + 8 * hi) * LDS_S + nT * 16 + lo;
#pragma unroll
            for (int j = 0; j < 8; ++j)
                srow[j * LDS_S] = (_Float16)acc[j];
        }
    }
    __syncthreads();

    // ---- fp32 causal softmax, one row per thread ----
    {
        const float scale = 0.125f;   // 1/sqrt(64)
        _Float16* srow = sS + t * LDS_S;
        float mx = -1e30f;
        for (int j = 0; j <= t; ++j)
            mx = fmaxf(mx, (float)srow[j] * scale);
        float sum = 0.f;
        for (int j = 0; j <= t; ++j) {
            float e = __expf((float)srow[j] * scale - mx);
            sum += e;
            srow[j] = (_Float16)e;
        }
        const float inv = 1.f / sum;
        for (int j = 0; j <= t; ++j)
            srow[j] = (_Float16)((float)srow[j] * inv);
        for (int j = t + 1; j < TT; ++j)
            srow[j] = (_Float16)0.f;  // zeroed P -> PV needs no predication
    }
    __syncthreads();

    // ---- O = P V: 16x64 per wave-tile-row, A fragment feeds 4 WMMAs ----
    for (int mT = wave * 2; mT < wave * 2 + 2; ++mT) {
        const _Float16* arow = sS + (mT * 16 + lo) * LDS_S;
        v8f acc0 = {}, acc1 = {}, acc2 = {}, acc3 = {};
#pragma unroll
        for (int k0 = 0; k0 < TT; k0 += 32) {
            v16h a = load_a_h16(arow + k0, hi);
            acc0 = wmma_f16(a, load_b_h16(sVt + ( 0 + lo) * LDVT + k0, hi), acc0);
            acc1 = wmma_f16(a, load_b_h16(sVt + (16 + lo) * LDVT + k0, hi), acc1);
            acc2 = wmma_f16(a, load_b_h16(sVt + (32 + lo) * LDVT + k0, hi), acc2);
            acc3 = wmma_f16(a, load_b_h16(sVt + (48 + lo) * LDVT + k0, hi), acc3);
        }
        _Float16* orow = attn + (size_t)(b * TT + mT * 16 + 8 * hi) * N_EMBD
                              + h * HSIZE + lo;
#pragma unroll
        for (int j = 0; j < 8; ++j) {
            orow[(size_t)j * N_EMBD +  0] = (_Float16)acc0[j];
            orow[(size_t)j * N_EMBD + 16] = (_Float16)acc1[j];
            orow[(size_t)j * N_EMBD + 32] = (_Float16)acc2[j];
            orow[(size_t)j * N_EMBD + 48] = (_Float16)acc3[j];
        }
    }
}

// ---------------------------------------------------------------------------
// Kernel 3: output projection out = attn @ Wo^T + bo (fp32), 16x64 per wave.
// ---------------------------------------------------------------------------
__global__ __launch_bounds__(256) void proj_kernel(
    const _Float16* __restrict__ attn,
    const _Float16* __restrict__ wo16,
    const float* __restrict__ bo,
    float* __restrict__ out) {

    const int lane = threadIdx.x & 31, wave = threadIdx.x >> 5;
    const int lo = lane & 15, hi = lane >> 4;

    const int cGroups = N_EMBD / 64;           // 6
    const int task = blockIdx.x * 8 + wave;    // 2048*6 = 12288 tasks
    const int mT = task / cGroups;
    const int cT = task - mT * cGroups;
    const int m0 = mT * 16, n0 = cT * 64;

    const _Float16* arow = attn + (size_t)(m0 + lo) * N_EMBD;
    const _Float16* brow = wo16 + (size_t)(n0 + lo) * N_EMBD;

    v8f acc0 = {}, acc1 = {}, acc2 = {}, acc3 = {};
#pragma unroll
    for (int k0 = 0; k0 < N_EMBD; k0 += 32) {
        if (k0 + 64 < N_EMBD) __builtin_prefetch(arow + k0 + 64, 0, 1);
        v16h a = load_a_h16(arow + k0, hi);
        acc0 = wmma_f16(a, load_b_h16(brow + k0, hi), acc0);
        acc1 = wmma_f16(a, load_b_h16(brow + (size_t)16 * N_EMBD + k0, hi), acc1);
        acc2 = wmma_f16(a, load_b_h16(brow + (size_t)32 * N_EMBD + k0, hi), acc2);
        acc3 = wmma_f16(a, load_b_h16(brow + (size_t)48 * N_EMBD + k0, hi), acc3);
    }

    float* orow = out + (size_t)(m0 + 8 * hi) * N_EMBD + n0 + lo;
    const float b0 = bo[n0 + lo], b1 = bo[n0 + 16 + lo];
    const float b2 = bo[n0 + 32 + lo], b3 = bo[n0 + 48 + lo];
#pragma unroll
    for (int j = 0; j < 8; ++j) {
        orow[(size_t)j * N_EMBD +  0] = acc0[j] + b0;
        orow[(size_t)j * N_EMBD + 16] = acc1[j] + b1;
        orow[(size_t)j * N_EMBD + 32] = acc2[j] + b2;
        orow[(size_t)j * N_EMBD + 48] = acc3[j] + b3;
    }
}

// ---------------------------------------------------------------------------
// Host launcher
// ---------------------------------------------------------------------------
extern "C" void kernel_launch(void* const* d_in, const int* in_sizes, int n_in,
                              void* d_out, int out_size, void* d_ws, size_t ws_size,
                              hipStream_t stream) {
    // setup_inputs() order: x, Wk, Wq, Wv, Wo, bo (all fp32)
    const float* x  = (const float*)d_in[0];
    const float* Wk = (const float*)d_in[1];
    const float* Wq = (const float*)d_in[2];
    const float* Wv = (const float*)d_in[3];
    const float* Wo = (const float*)d_in[4];
    const float* bo = (const float*)d_in[5];
    float* out = (float*)d_out;

    // Workspace (all fp16, L2-resident: ~127MB < 192MB L2):
    //   qkv[32768x1152] | attn[32768x384] | x16[32768x384] | w16[1152x384] | wo16[384x384]
    _Float16* qkv  = (_Float16*)d_ws;
    _Float16* attn = qkv  + (size_t)M_TOTAL * QKV_N;
    _Float16* x16  = attn + (size_t)M_TOTAL * N_EMBD;
    _Float16* w16  = x16  + (size_t)M_TOTAL * N_EMBD;
    _Float16* wo16 = w16  + (size_t)QKV_N * N_EMBD;

    // Kernel 0: (12582912 + 442368 + 147456)/4 quads / 256 = 12864 blocks exactly
    cvt_kernel<<<12864, 256, 0, stream>>>(x, Wq, Wk, Wv, Wo, x16, w16, wo16);

    // Kernel 1: 36864 wave-tasks / 8 waves = 4608 blocks
    qkv_kernel<<<4608, 256, 0, stream>>>(x16, w16, qkv);

    // Kernel 2: one WG per (b,h); 237KB dynamic LDS (gfx1250: 320KB/WGP)
    const size_t smemBytes =
        (size_t)(2 * TT * LDQ + HSIZE * LDVT + TT * LDS_S) * sizeof(_Float16); // 242688
    (void)hipFuncSetAttribute((const void*)attn_kernel,
                              hipFuncAttributeMaxDynamicSharedMemorySize,
                              (int)smemBytes);
    attn_kernel<<<BB * NHEAD, 256, smemBytes, stream>>>(qkv, attn);

    // Kernel 3: 12288 wave-tasks / 8 waves = 1536 blocks
    proj_kernel<<<1536, 256, 0, stream>>>(attn, wo16, bo, out);
}
